// MSA_28329604284992
// MI455X (gfx1250) — matrix operand order repaired and checked by reading, not compiled, not run on hardware
//
#include <hip/hip_runtime.h>

#define Bc 16
#define Nc 1024
#define Dc 768
#define Hc 12
#define DHc 64

typedef __attribute__((ext_vector_type(16))) _Float16 v16h;
typedef __attribute__((ext_vector_type(8)))  _Float16 v8h;
typedef __attribute__((ext_vector_type(8)))  float    v8f;

static __device__ __forceinline__ v16h cat16(v8h lo, v8h hi) {
  return __builtin_shufflevector(lo, hi, 0,1,2,3,4,5,6,7,8,9,10,11,12,13,14,15);
}

static __device__ __forceinline__ v8f wmma_f16(v16h a, v16h b, v8f c) {
  // D = A(16x32 f16) x B(32x16 f16) + C(16x16 f32)
  return __builtin_amdgcn_wmma_f32_16x16x32_f16(false, a, false, b, (short)0, c, false, false);
}

// CDNA5 async global->LDS copy (ASYNCcnt-tracked, no VGPR data path).
// vdst = per-lane LDS byte address (low 32 bits of generic shared pointer),
// vaddr = per-lane 64-bit global address. GV addressing mode.
static __device__ __forceinline__ void async_ld_b128(unsigned lds_addr, const void* g) {
  asm volatile("global_load_async_to_lds_b128 %0, %1, off"
               :: "v"(lds_addr), "v"((unsigned long long)(size_t)g)
               : "memory");
}
#define WAIT_ASYNCCNT(n) asm volatile("s_wait_asynccnt %0" :: "i"(n) : "memory")

// ---------------------------------------------------------------------------
// Kernel 1: per-head QKV projection. x[B,N,768] f32 -> Q[BH,N,64] f16,
// K[BH,N,64] f16, Vt[BH,64,N] f16.  y = x @ W^T + b per head.
// Block = (b,h, 128-row slab); 8 waves x 16 rows each.
// ---------------------------------------------------------------------------
__global__ __launch_bounds__(256) void qkv_kernel(
    const float* __restrict__ x,
    const float* __restrict__ Wq, const float* __restrict__ Wk, const float* __restrict__ Wv,
    const float* __restrict__ bq, const float* __restrict__ bk, const float* __restrict__ bv,
    _Float16* __restrict__ Qg, _Float16* __restrict__ Kg, _Float16* __restrict__ Vtg)
{
  const int slab = blockIdx.x & 7;
  const int bh   = blockIdx.x >> 3;
  const int h    = bh % Hc;
  const int b    = bh / Hc;
  const int wave = threadIdx.x >> 5;
  const int lane = threadIdx.x & 31;
  const int col  = lane & 15;
  const int hiw  = lane >> 4;
  const int n0   = slab * 128 + wave * 16;
  const int kb8  = hiw ? 8 : 0;    // A-fragment K base per half-wave
  const int kb16 = hiw ? 16 : 0;   // unused; B fragments also use kb8

  // --- A fragments from x (row = query n, K = feature d), f32 -> f16 -------
  const float* xrow = x + ((size_t)b * Nc + (n0 + col)) * Dc + h * DHc;
  v16h xa[2];
#pragma unroll
  for (int c = 0; c < 2; ++c) {
    const float* p = xrow + 32 * c + kb8;
    v8f f0 = *(const v8f*)(p);        // K = kb8 .. kb8+7
    v8f f1 = *(const v8f*)(p + 16);   // K = kb8+16 .. kb8+23
    v16h a;
#pragma unroll
    for (int u = 0; u < 8; ++u) { a[u] = (_Float16)f0[u]; a[u + 8] = (_Float16)f1[u]; }
    xa[c] = a;
  }

  const float* Ws[3] = { Wq, Wk, Wv };
  const float* bs[3] = { bq, bk, bv };

#pragma unroll
  for (int mat = 0; mat < 3; ++mat) {
    const float* W    = Ws[mat];
    const float* bias = bs[mat];
#pragma unroll
    for (int et = 0; et < 4; ++et) {
      const int e = et * 16 + col;          // output feature (C-layout column)
      const float bval = bias[h * DHc + e];
      v8f acc;
#pragma unroll
      for (int r = 0; r < 8; ++r) acc[r] = bval;   // bias broadcast over rows
#pragma unroll
      for (int c = 0; c < 2; ++c) {
        // B element (k=d, n=e) = W[e][d]; per lane: d = kb8..kb8+7, kb8+16..kb8+23
        const float* wp = W + ((size_t)h * DHc + e) * DHc + 32 * c + kb8;
        v8f w0 = *(const v8f*)(wp);
        v8f w1 = *(const v8f*)(wp + 16);
        v16h bf;
#pragma unroll
        for (int u = 0; u < 8; ++u) { bf[u] = (_Float16)w0[u]; bf[u + 8] = (_Float16)w1[u]; }
        acc = wmma_f16(xa[c], bf, acc);
      }
      if (mat < 2) {
        // Q / K natural [n][e] layout (feeds A-fragments / K LDS tiles)
        _Float16* dst = (mat == 0) ? Qg : Kg;
#pragma unroll
        for (int r = 0; r < 8; ++r) {
          const int m = r + 8 * hiw;
          dst[((size_t)bh * Nc + n0 + m) * DHc + e] = (_Float16)acc[r];
        }
      } else {
        // V transposed [e][n]: lane's 8 rows are contiguous -> 16B store
        v8h pk;
#pragma unroll
        for (int r = 0; r < 8; ++r) pk[r] = (_Float16)acc[r];
        *(v8h*)(Vtg + ((size_t)bh * DHc + e) * Nc + n0 + 8 * hiw) = pk;
      }
    }
  }
}

// ---------------------------------------------------------------------------
// Kernel 2: flash attention per (b,h) over 128 query rows / block.
// 8 waves x 16 rows; KV streamed in 64-key tiles, double-buffered in LDS via
// CDNA5 async global->LDS loads overlapped with WMMA compute.
// ---------------------------------------------------------------------------
__global__ __launch_bounds__(256) void attn_kernel(
    const _Float16* __restrict__ Qg,
    const _Float16* __restrict__ Kg,
    const _Float16* __restrict__ Vtg,
    float* __restrict__ out)
{
  constexpr int KS = 72;                      // padded LDS row stride (halves)
  constexpr int NT = Nc / 64;                 // 16 key tiles
  __shared__ _Float16 KL[2][64 * KS];         // K tile  [j][e], double-buffered
  __shared__ _Float16 VL[2][64 * KS];         // V tile  [e][j], double-buffered
  __shared__ _Float16 PL[8 * 16 * KS];        // per-wave P tile [i][j]

  const int slab = blockIdx.x & 7;
  const int bh   = blockIdx.x >> 3;
  const int h    = bh % Hc;
  const int b    = bh / Hc;
  const int wave = threadIdx.x >> 5;
  const int lane = threadIdx.x & 31;
  const int col  = lane & 15;
  const int hiw  = lane >> 4;
  const int n0   = slab * 128 + wave * 16;
  const int kb8  = hiw ? 8 : 0;
  const int kb16 = hiw ? 16 : 0;

  const int ldrow = threadIdx.x >> 2;         // 0..63
  const int ldseg = (threadIdx.x & 3) * 16;   // 0,16,32,48

  // issue one KV tile's async loads: 4 instr/wave, 16B/lane each
  auto issue_tile = [&](int kt, int buf) {
    const int j0 = kt * 64;
    const _Float16* ks = Kg + ((size_t)bh * Nc + j0 + ldrow) * DHc + ldseg;
    const unsigned ka = (unsigned)(size_t)&KL[buf][ldrow * KS + ldseg];
    async_ld_b128(ka,      ks);
    async_ld_b128(ka + 16, ks + 8);
    const _Float16* vs = Vtg + ((size_t)bh * DHc + ldrow) * Nc + j0 + ldseg;
    const unsigned va = (unsigned)(size_t)&VL[buf][ldrow * KS + ldseg];
    async_ld_b128(va,      vs);
    async_ld_b128(va + 16, vs + 8);
  };

  // --- Q A-fragments, kept in registers for all 16 KV tiles ----------------
  const _Float16* qrow = Qg + ((size_t)bh * Nc + n0 + col) * DHc;
  v16h qa[2];
#pragma unroll
  for (int c = 0; c < 2; ++c)
    qa[c] = cat16(*(const v8h*)(qrow + 32 * c + kb8),
                  *(const v8h*)(qrow + 32 * c + kb8 + 16));

  v8f o[4];
  float mr[8], lr[8];
#pragma unroll
  for (int et = 0; et < 4; ++et)
#pragma unroll
    for (int r = 0; r < 8; ++r) o[et][r] = 0.0f;
#pragma unroll
  for (int r = 0; r < 8; ++r) { mr[r] = -1.0e30f; lr[r] = 0.0f; }

  issue_tile(0, 0);                           // prefetch first tile

  for (int kt = 0; kt < NT; ++kt) {
    const int buf = kt & 1;
    if (kt + 1 < NT) {
      issue_tile(kt + 1, buf ^ 1);            // overlap next tile with compute
      WAIT_ASYNCCNT(4);                       // in-order: current tile landed
    } else {
      WAIT_ASYNCCNT(0);
    }
    __syncthreads();                          // tile visible to all waves

    // --- S = Q K^T (4 tiles of 16x16, f32) ---------------------------------
    v8f sacc[4];
#pragma unroll
    for (int t = 0; t < 4; ++t) {
      const _Float16* kp = &KL[buf][(t * 16 + col) * KS];  // B elem (k=e, n=j)
      v8f s;
#pragma unroll
      for (int r = 0; r < 8; ++r) s[r] = 0.0f;
#pragma unroll
      for (int c = 0; c < 2; ++c) {
        v16h bf = cat16(*(const v8h*)(kp + 32 * c + kb8),
                        *(const v8h*)(kp + 32 * c + kb8 + 16));
        s = wmma_f16(qa[c], bf, s);
      }
      sacc[t] = s;
    }
#pragma unroll
    for (int t = 0; t < 4; ++t)
#pragma unroll
      for (int r = 0; r < 8; ++r) sacc[t][r] *= 0.125f;   // 1/sqrt(64)

    // --- online softmax: row max / correction / P / row sum ----------------
    float mnew[8], cf[8];
#pragma unroll
    for (int r = 0; r < 8; ++r) {
      float v = fmaxf(fmaxf(sacc[0][r], sacc[1][r]), fmaxf(sacc[2][r], sacc[3][r]));
#pragma unroll
      for (int msk = 1; msk < 16; msk <<= 1) v = fmaxf(v, __shfl_xor(v, msk, 32));
      mnew[r] = fmaxf(mr[r], v);
      cf[r]   = __expf(mr[r] - mnew[r]);
      mr[r]   = mnew[r];
    }
    _Float16* Pw = PL + wave * 16 * KS;
    float rsum[8];
#pragma unroll
    for (int r = 0; r < 8; ++r) rsum[r] = 0.0f;
#pragma unroll
    for (int t = 0; t < 4; ++t)
#pragma unroll
      for (int r = 0; r < 8; ++r) {
        const float p = __expf(sacc[t][r] - mnew[r]);
        Pw[(r + 8 * hiw) * KS + t * 16 + col] = (_Float16)p;   // [i][j]
        rsum[r] += p;
      }
#pragma unroll
    for (int r = 0; r < 8; ++r) {
      float v = rsum[r];
#pragma unroll
      for (int msk = 1; msk < 16; msk <<= 1) v += __shfl_xor(v, msk, 32);
      lr[r] = lr[r] * cf[r] + v;
#pragma unroll
      for (int et = 0; et < 4; ++et) o[et][r] *= cf[r];
    }

    // --- O += P V : reload P as A-fragments (LDS in-order per wave) --------
    const _Float16* prow = Pw + col * KS;
    v16h pa[2];
#pragma unroll
    for (int c = 0; c < 2; ++c)
      pa[c] = cat16(*(const v8h*)(prow + 32 * c + kb8),
                    *(const v8h*)(prow + 32 * c + kb8 + 16));
#pragma unroll
    for (int et = 0; et < 4; ++et) {
      const _Float16* vp = &VL[buf][(et * 16 + col) * KS];  // B elem (k=j, n=e)
#pragma unroll
      for (int c = 0; c < 2; ++c) {
        v16h bf = cat16(*(const v8h*)(vp + 32 * c + kb8),
                        *(const v8h*)(vp + 32 * c + kb8 + 16));
        o[et] = wmma_f16(pa[c], bf, o[et]);
      }
    }
    __syncthreads();  // all waves done with buf before it is refilled
  }

  // --- epilogue: divide by row sum, store f32 into [B,N,768] ---------------
#pragma unroll
  for (int et = 0; et < 4; ++et)
#pragma unroll
    for (int r = 0; r < 8; ++r) {
      const int m = r + 8 * hiw;
      out[((size_t)b * Nc + n0 + m) * Dc + h * DHc + et * 16 + col] = o[et][r] / lr[r];
    }
}

// ---------------------------------------------------------------------------
extern "C" void kernel_launch(void* const* d_in, const int* in_sizes, int n_in,
                              void* d_out, int out_size, void* d_ws, size_t ws_size,
                              hipStream_t stream) {
  (void)in_sizes; (void)n_in; (void)out_size; (void)ws_size;
  const float* x  = (const float*)d_in[0];
  const float* Wq = (const float*)d_in[1];
  const float* Wk = (const float*)d_in[2];
  const float* Wv = (const float*)d_in[3];
  const float* bq = (const float*)d_in[4];
  const float* bk = (const float*)d_in[5];
  const float* bv = (const float*)d_in[6];
  float* out = (float*)d_out;

  const size_t per = (size_t)Bc * Hc * Nc * DHc;   // 12.58M f16 elements each
  _Float16* Qg  = (_Float16*)d_ws;
  _Float16* Kg  = Qg + per;
  _Float16* Vtg = Kg + per;

  const dim3 blk(256);
  const dim3 grid(Bc * Hc * (Nc / 128));           // 1536 blocks
  qkv_kernel<<<grid, blk, 0, stream>>>(x, Wq, Wk, Wv, bq, bk, bv, Qg, Kg, Vtg);
  attn_kernel<<<grid, blk, 0, stream>>>(Qg, Kg, Vtg, out);
}
